// LeafColorGraphModel_7808250544741
// MI455X (gfx1250) — matrix operand order repaired and checked by reading, not compiled, not run on hardware
//
#include <hip/hip_runtime.h>
#include <stdint.h>

// ---------------------------------------------------------------------------
// LeafColorGraphModel on MI455X (gfx1250, wave32, WMMA)
// GEMMs: v_wmma_f32_16x16x32_bf16 with 3xBF16 split precision (hi/lo planes
// pre-packed in WMMA register-pair order; fragments = aligned b128 LDS loads).
// W tiles staged global->LDS with global_load_async_to_lds_b128 (ASYNCcnt).
// ---------------------------------------------------------------------------

#define TREE_TOTAL 2047
#define NG 64
#define NN (NG * TREE_TOTAL)   // 131008 nodes
#define NE (2 * NN)            // 262016 edges
#define HD 256
#define NLAYERS 4
#define BN_EPS 1e-5f
#define LN_EPS 1e-5f

#define BM 64
#define BK 32
#define NPAIR (BK / 2)         // 16 packed K-pairs per row per chunk
#define LDS_STRIDE (NPAIR + 4) // pad 4 dwords: keeps 16B align, shifts banks
#define NCHUNK (HD / BK)       // 8 K-chunks
#define WPLANE (NCHUNK * HD * NPAIR)   // dwords per W plane (8*256*16 = 32768)

typedef __attribute__((ext_vector_type(16))) __bf16 v16bf;
typedef __attribute__((ext_vector_type(8)))  float  v8f;
typedef int i32x4 __attribute__((vector_size(16)));   // async builtin ptr type

struct __attribute__((aligned(16))) U4 { uint32_t a, b, c, d; };

union Frag {
    v16bf    v;
    U4       q[2];
};

#if __has_builtin(__builtin_amdgcn_global_load_async_to_lds_b128) && \
    __has_builtin(__builtin_amdgcn_s_wait_asynccnt)
#define HAVE_ASYNC_LDS 1
#else
#define HAVE_ASYNC_LDS 0
#endif

__device__ __forceinline__ uint32_t bf16_rne(float x) {
    union { float f; uint32_t u; } c; c.f = x;
    return (c.u + 0x7fffu + ((c.u >> 16) & 1u)) >> 16;
}
__device__ __forceinline__ float bf16_to_f32(uint32_t h) {
    union { uint32_t u; float f; } c; c.u = h << 16;
    return c.f;
}
// split x0,x1 into packed hi pair and packed lo (residual) pair
__device__ __forceinline__ void split_pair(float v0, float v1,
                                           uint32_t& hi, uint32_t& lo) {
    uint32_t h0 = bf16_rne(v0), h1 = bf16_rne(v1);
    uint32_t l0 = bf16_rne(v0 - bf16_to_f32(h0));
    uint32_t l1 = bf16_rne(v1 - bf16_to_f32(h1));
    hi = (h0 & 0xffffu) | (h1 << 16);
    lo = (l0 & 0xffffu) | (l1 << 16);
}

// ---------------------------------------------------------------------------
// generic zero
// ---------------------------------------------------------------------------
__global__ __launch_bounds__(256) void zero_kernel(float* __restrict__ p, long n) {
    long t = (long)blockIdx.x * 256 + threadIdx.x;
    if (t < n) p[t] = 0.f;
}

// ---------------------------------------------------------------------------
// embedding gather: h[n] = concat(emb_node[x[n,0]], emb_label[x[n,1]])
// ---------------------------------------------------------------------------
__global__ __launch_bounds__(256) void embed_kernel(
    const int* __restrict__ x,
    const float* __restrict__ emb_node,
    const float* __restrict__ emb_label,
    float* __restrict__ h)
{
    long t = (long)blockIdx.x * 256 + threadIdx.x;
    if (t >= (long)NN * HD) return;
    int node = (int)(t >> 8);
    int f = (int)(t & 255);
    float v;
    if (f < 128) v = emb_node[x[2 * node] * 128 + f];
    else         v = emb_label[x[2 * node + 1] * 128 + (f - 128)];
    h[t] = v;
}

// ---------------------------------------------------------------------------
// W pre-split: f32 [HD,HD] -> packed bf16 hi/lo planes in per-chunk tile order
// plane layout (dwords): plane*WPLANE + kc*HD*NPAIR + j*NPAIR + pp
// ---------------------------------------------------------------------------
__global__ __launch_bounds__(256) void wprep_kernel(
    const float* __restrict__ W, uint32_t* __restrict__ Wp)
{
    int t = blockIdx.x * 256 + threadIdx.x;   // HD*128 = 32768 threads
    int j = t >> 7;
    int p = t & 127;                          // K-pair index, k = 2p
    int kc = p >> 4;                          // chunk
    int pp = p & 15;
    float2 w2 = *(const float2*)(W + j * HD + 2 * p);
    uint32_t hi, lo;
    split_pair(w2.x, w2.y, hi, lo);
    long base = ((long)kc * HD + j) * NPAIR + pp;
    Wp[base] = hi;
    Wp[WPLANE + base] = lo;
}

// ---------------------------------------------------------------------------
// edge aggregation: agg[dst] += h[src] * w  (float atomics)
// ---------------------------------------------------------------------------
__global__ __launch_bounds__(256) void edge_agg_kernel(
    const float* __restrict__ h,
    const int* __restrict__ src,
    const int* __restrict__ dst,
    const float* __restrict__ w,
    float* __restrict__ agg)
{
    long t = (long)blockIdx.x * 256 + threadIdx.x;
    int e = (int)(t >> 6);
    int c = (int)(t & 63) * 4;
    if (e >= NE) return;
    int s = src[e], d = dst[e];
    float wt = w[e];
    const float4 hv = *(const float4*)(h + (long)s * HD + c);
    float* ap = agg + (long)d * HD + c;
    atomicAdd(ap + 0, hv.x * wt);
    atomicAdd(ap + 1, hv.y * wt);
    atomicAdd(ap + 2, hv.z * wt);
    atomicAdd(ap + 3, hv.w * wt);
}

// ---------------------------------------------------------------------------
// last layer: every node of a graph adds h into its root row (deterministic)
// ---------------------------------------------------------------------------
__global__ __launch_bounds__(64) void root_agg_kernel(
    const float* __restrict__ h,
    const int* __restrict__ root_index,
    float* __restrict__ agg)
{
    int g = blockIdx.x;
    int f = blockIdx.y * 64 + threadIdx.x;
    long base = (long)g * TREE_TOTAL;
    float s = 0.f;
    for (int t = 0; t < TREE_TOTAL; ++t)
        s += h[(base + t) * HD + f];
    agg[(long)root_index[g] * HD + f] = s;
}

// ---------------------------------------------------------------------------
// fused GEMM (WMMA bf16 split precision) + bias + BN column statistics.
//   mode 0: in = A + A2                       (h + agg)
//   mode 1: in = relu(A*inScale + inShift)    (BN1+relu of t1)
// Out = in @ W.T + bias ; atomic col sums / sumsq for batchnorm.
// ---------------------------------------------------------------------------
__global__ __launch_bounds__(256) void gemm_bn_stats_wmma(
    const float* __restrict__ A,
    const float* __restrict__ A2,
    const float* __restrict__ inScale,
    const float* __restrict__ inShift,
    const uint32_t* __restrict__ Wp,   // pre-split planes (wprep layout)
    const float* __restrict__ bias,    // [HD]
    float* __restrict__ Out,           // [NN, HD]
    float* __restrict__ sums,          // [HD]
    float* __restrict__ sumsq,         // [HD]
    int mode)
{
    // plane 0 = hi, plane 1 = lo; rows padded to LDS_STRIDE dwords
    __shared__ __align__(16) uint32_t sA[2][BM][LDS_STRIDE];
    __shared__ __align__(16) uint32_t sB[2][HD][LDS_STRIDE];

    const int tid = threadIdx.x;
    const int block_row = blockIdx.x * BM;
    const int wave = tid >> 5;
    const int lane = tid & 31;
    const int row_tile = wave & 3;      // 4 row tiles of 16
    const int col_half = wave >> 2;     // waves 0-3: cols 0..127, 4-7: 128..255
    const int ln16 = lane & 15;
    const int lhi = lane >> 4;          // 0: lanes 0-15, 1: lanes 16-31

    v8f acc[8] = {};

    for (int kcIdx = 0; kcIdx < NCHUNK; ++kcIdx) {
        const int kc = kcIdx * BK;
        __syncthreads();   // previous compute done before LDS overwrite

        // ---- stage W tiles (both planes): 2*HD*NPAIR dwords = 2048 U4 / 256 thr
        {
            const U4* gW = (const U4*)(Wp + (long)kcIdx * HD * NPAIR);
            const U4* gWlo = (const U4*)(Wp + WPLANE + (long)kcIdx * HD * NPAIR);
            #pragma unroll
            for (int r = 0; r < 8; ++r) {
                int q = tid + 256 * r;          // 0..2047
                int plane = q >> 10;            // 0 hi / 1 lo
                int idx = q & 1023;             // U4 index within plane
                int j = idx >> 2, pp = (idx & 3) * 4;
                const U4* gsrc = (plane ? gWlo : gW) + idx;
                uint32_t* ldst = &sB[plane][j][pp];
#if HAVE_ASYNC_LDS
                __builtin_amdgcn_global_load_async_to_lds_b128(
                    (i32x4*)gsrc, (i32x4*)ldst, 0, 0);
#else
                *(U4*)ldst = *gsrc;
#endif
            }
        }

        // ---- stage A tile: BM*NPAIR pairs = 1024 / 256 thr = 4 each
        #pragma unroll
        for (int r = 0; r < 4; ++r) {
            int p = tid + 256 * r;
            int row = p >> 4, pp = p & 15;
            int col = kc + 2 * pp;
            long gi = (long)(block_row + row) * HD + col;
            float v0, v1;
            if (mode == 0) {
                float2 a = *(const float2*)(A + gi);
                float2 b = *(const float2*)(A2 + gi);
                v0 = a.x + b.x; v1 = a.y + b.y;
            } else {
                float2 a = *(const float2*)(A + gi);
                v0 = fmaxf(a.x * inScale[col] + inShift[col], 0.f);
                v1 = fmaxf(a.y * inScale[col + 1] + inShift[col + 1], 0.f);
            }
            uint32_t hi, lo;
            split_pair(v0, v1, hi, lo);
            sA[0][row][pp] = hi;
            sA[1][row][pp] = lo;
        }

#if HAVE_ASYNC_LDS
        __builtin_amdgcn_s_wait_asynccnt(0);
#endif
        __syncthreads();

        // ---- A fragments: two aligned b128 loads per plane
        const int m = 16 * row_tile + ln16;
        Frag Ahi, Alo;
        Ahi.q[0] = *(const U4*)&sA[0][m][lhi * 4];
        Ahi.q[1] = *(const U4*)&sA[0][m][8 + lhi * 4];
        Alo.q[0] = *(const U4*)&sA[1][m][lhi * 4];
        Alo.q[1] = *(const U4*)&sA[1][m][8 + lhi * 4];

        // ---- per column tile: B fragments are pure b128 loads, then 3 WMMA
        #pragma unroll
        for (int jt = 0; jt < 8; ++jt) {
            int col = 16 * (8 * col_half + jt) + ln16;
            Frag Bhi, Blo;
            Bhi.q[0] = *(const U4*)&sB[0][col][lhi * 4];
            Bhi.q[1] = *(const U4*)&sB[0][col][8 + lhi * 4];
            Blo.q[0] = *(const U4*)&sB[1][col][lhi * 4];
            Blo.q[1] = *(const U4*)&sB[1][col][8 + lhi * 4];
            // D += Ahi*Bhi + Ahi*Blo + Alo*Bhi  (3xBF16 ~ fp32)
            acc[jt] = __builtin_amdgcn_wmma_f32_16x16x32_bf16(
                false, Ahi.v, false, Bhi.v, (short)0, acc[jt], false, false);
            acc[jt] = __builtin_amdgcn_wmma_f32_16x16x32_bf16(
                false, Ahi.v, false, Blo.v, (short)0, acc[jt], false, false);
            acc[jt] = __builtin_amdgcn_wmma_f32_16x16x32_bf16(
                false, Alo.v, false, Bhi.v, (short)0, acc[jt], false, false);
        }
    }

    // ---- epilogue: bias, store, column stats (VGPR r -> row r + 8*lhi)
    #pragma unroll
    for (int jt = 0; jt < 8; ++jt) {
        int col = 16 * (8 * col_half + jt) + ln16;
        float b = bias[col];
        float cs = 0.f, cq = 0.f;
        #pragma unroll
        for (int r = 0; r < 8; ++r) {
            float v = acc[jt][r] + b;
            int grow = block_row + 16 * row_tile + r + 8 * lhi;
            Out[(long)grow * HD + col] = v;
            cs += v;
            cq += v * v;
        }
        atomicAdd(&sums[col], cs);
        atomicAdd(&sumsq[col], cq);
    }
}

// ---------------------------------------------------------------------------
// BN finalize: scale/shift from accumulated stats; reset stats.
// ---------------------------------------------------------------------------
__global__ __launch_bounds__(256) void bn_finalize_kernel(
    float* __restrict__ sums, float* __restrict__ sumsq,
    const float* __restrict__ g, const float* __restrict__ be,
    float* __restrict__ scale, float* __restrict__ shift, float inv_n)
{
    int f = threadIdx.x;
    float m = sums[f] * inv_n;
    float v = sumsq[f] * inv_n - m * m;
    float s = g[f] * rsqrtf(v + BN_EPS);
    scale[f] = s;
    shift[f] = be[f] - m * s;
    sums[f] = 0.f;
    sumsq[f] = 0.f;
}

// ---------------------------------------------------------------------------
// fused BN2 + relu + residual + LayerNorm: one wave per node row.
// ---------------------------------------------------------------------------
__global__ __launch_bounds__(256) void bn_relu_res_ln_kernel(
    const float* __restrict__ t2,
    const float* __restrict__ scale, const float* __restrict__ shift,
    const float* __restrict__ ln_g, const float* __restrict__ ln_b,
    float* __restrict__ h)
{
    int wave = threadIdx.x >> 5;
    int lane = threadIdx.x & 31;
    int row = blockIdx.x * 8 + wave;
    if (row >= NN) return;
    long base = (long)row * HD;
    float r[8];
    float s = 0.f;
    #pragma unroll
    for (int i = 0; i < 8; ++i) {
        int f = lane + 32 * i;
        float v = t2[base + f] * scale[f] + shift[f];
        v = v > 0.f ? v : 0.f;
        float x = h[base + f] + v;           // residual
        r[i] = x;
        s += x;
    }
    #pragma unroll
    for (int off = 16; off > 0; off >>= 1) s += __shfl_xor(s, off, 32);
    float mean = s * (1.f / HD);
    float q = 0.f;
    #pragma unroll
    for (int i = 0; i < 8; ++i) { float d = r[i] - mean; q += d * d; }
    #pragma unroll
    for (int off = 16; off > 0; off >>= 1) q += __shfl_xor(q, off, 32);
    float rs = rsqrtf(q * (1.f / HD) + LN_EPS);
    #pragma unroll
    for (int i = 0; i < 8; ++i) {
        int f = lane + 32 * i;
        h[base + f] = ln_g[f] * (r[i] - mean) * rs + ln_b[f];
    }
}

// ---------------------------------------------------------------------------
// readout: out[g,o] = h[root[g]] . Wout[o]  (64x32x256, tiny)
// ---------------------------------------------------------------------------
__global__ __launch_bounds__(256) void readout_kernel(
    const float* __restrict__ h,
    const int* __restrict__ root_index,
    const float* __restrict__ Wout,
    float* __restrict__ out)
{
    __shared__ float red[256];
    int g = blockIdx.x;
    int o = threadIdx.x & 31;
    int part = threadIdx.x >> 5;
    long base = (long)root_index[g] * HD;
    float s = 0.f;
    for (int k = part * 32; k < part * 32 + 32; ++k)
        s += h[base + k] * Wout[o * HD + k];
    red[threadIdx.x] = s;
    __syncthreads();
    if (part == 0) {
        float t = 0.f;
        #pragma unroll
        for (int p = 0; p < 8; ++p) t += red[o + 32 * p];
        out[g * 32 + o] = t;
    }
}

// ---------------------------------------------------------------------------
// launch
// ---------------------------------------------------------------------------
extern "C" void kernel_launch(void* const* d_in, const int* in_sizes, int n_in,
                              void* d_out, int out_size, void* d_ws, size_t ws_size,
                              hipStream_t stream) {
    const int*   x          = (const int*)d_in[0];
    const int*   edge_src   = (const int*)d_in[1];
    const int*   edge_dst   = (const int*)d_in[2];
    const float* edge_wt    = (const float*)d_in[3];
    // d_in[4] = batch (implicit: node/TREE_TOTAL), unused
    const int*   root_index = (const int*)d_in[5];
    const float* emb_node   = (const float*)d_in[6];
    const float* emb_label  = (const float*)d_in[7];
    const float* W1  = (const float*)d_in[8];
    const float* b1  = (const float*)d_in[9];
    const float* g1  = (const float*)d_in[10];
    const float* be1 = (const float*)d_in[11];
    const float* W2  = (const float*)d_in[12];
    const float* b2  = (const float*)d_in[13];
    const float* g2  = (const float*)d_in[14];
    const float* be2 = (const float*)d_in[15];
    const float* ln_g = (const float*)d_in[16];
    const float* ln_b = (const float*)d_in[17];
    const float* Wout = (const float*)d_in[18];
    float* out = (float*)d_out;

    const long NH = (long)NN * HD;
    float* ws   = (float*)d_ws;
    float* h    = ws;               // [NN,HD]
    float* bufA = ws + NH;          // agg / t2
    float* bufB = ws + 2 * NH;      // t1
    float* stats = ws + 3 * NH;
    float* sums  = stats;           // [256]
    float* sumsq = stats + 256;     // [256]
    float* sc1   = stats + 512;
    float* sh1   = stats + 768;
    float* sc2   = stats + 1024;
    float* sh2   = stats + 1280;
    uint32_t* wp1 = (uint32_t*)(stats + 2048);            // 2*WPLANE dwords
    uint32_t* wp2 = wp1 + 2 * WPLANE;

    const float inv_n = 1.f / (float)NN;

    // init
    zero_kernel<<<2, 256, 0, stream>>>(sums, 512);
    embed_kernel<<<(unsigned)(NH / 256), 256, 0, stream>>>(x, emb_node, emb_label, h);

    for (int i = 0; i < NLAYERS; ++i) {
        // pre-split this layer's weights into bf16 hi/lo tile planes
        wprep_kernel<<<128, 256, 0, stream>>>(W1 + (long)i * HD * HD, wp1);
        wprep_kernel<<<128, 256, 0, stream>>>(W2 + (long)i * HD * HD, wp2);

        // aggregation into bufA
        zero_kernel<<<(unsigned)(NH / 256), 256, 0, stream>>>(bufA, NH);
        if (i < NLAYERS - 1) {
            edge_agg_kernel<<<(unsigned)((long)NE * 64 / 256), 256, 0, stream>>>(
                h, edge_src, edge_dst, edge_wt, bufA);
        } else {
            dim3 grid(NG, 4);
            root_agg_kernel<<<grid, 64, 0, stream>>>(h, root_index, bufA);
        }
        // t1 = (h+agg) @ W1.T + b1  (+ BN1 stats)
        gemm_bn_stats_wmma<<<NN / BM, 256, 0, stream>>>(
            h, bufA, nullptr, nullptr, wp1, b1 + i * HD, bufB, sums, sumsq, 0);
        bn_finalize_kernel<<<1, 256, 0, stream>>>(sums, sumsq, g1 + i * HD, be1 + i * HD,
                                                  sc1, sh1, inv_n);
        // t2 = relu(BN1(t1)) @ W2.T + b2  (+ BN2 stats)
        gemm_bn_stats_wmma<<<NN / BM, 256, 0, stream>>>(
            bufB, nullptr, sc1, sh1, wp2, b2 + i * HD, bufA, sums, sumsq, 1);
        bn_finalize_kernel<<<1, 256, 0, stream>>>(sums, sumsq, g2 + i * HD, be2 + i * HD,
                                                  sc2, sh2, inv_n);
        // h = LayerNorm(h + relu(BN2(t2)))
        bn_relu_res_ln_kernel<<<NN / 8, 256, 0, stream>>>(bufA, sc2, sh2,
                                                          ln_g + i * HD, ln_b + i * HD, h);
    }

    readout_kernel<<<NG, 256, 0, stream>>>(h, root_index, Wout, out);
}
